// HeteroGraphSAGE_42803644071962
// MI455X (gfx1250) — matrix operand order repaired and checked by reading, not compiled, not run on hardware
//
#include <hip/hip_runtime.h>

// ---------------------------------------------------------------------------
// HeteroGraphSAGE (LSTM aggregator) for MI455X / gfx1250, bf16 WMMA path.
// N0=131072, N1=16384, N2=4096, K=8, D=H=256.
// ---------------------------------------------------------------------------

typedef __attribute__((ext_vector_type(16))) __bf16 v16bf;
typedef __attribute__((ext_vector_type(8)))  __bf16 v8bf;
typedef __attribute__((ext_vector_type(8)))  float  v8f;

#define WMMA_BF16(A, B, C) \
  __builtin_amdgcn_wmma_f32_16x16x32_bf16(false, (A), false, (B), (short)0, (C), false, false)

static __device__ __forceinline__ float sigf(float x) { return 1.0f / (1.0f + __expf(-x)); }
static __device__ __forceinline__ float geluf(float x) {
  return 0.5f * x * (1.0f + erff(x * 0.7071067811865475f));
}

union V16U { v16bf v; v8bf h[2]; };

// A fragment (16x32 bf16) from an LDS row-major [64][256] bf16 buffer.
// lane<16: row=tile+lane, K = kc*32 + {0..7, 16..23}; lane>=16: K = {8..15, 24..31}.
static __device__ __forceinline__ v16bf load_a(const __bf16* base, int row, int kc, int hi) {
  const __bf16* p = base + row * 256 + kc * 32 + hi * 8;
  V16U u;
  u.h[0] = *(const v8bf*)p;
  u.h[1] = *(const v8bf*)(p + 16);
  return u.v;
}

// B fragment (32x16 bf16) from the packed global weight array (KC = 256/32 = 8).
// lane l holds 16 contiguous bf16: W[jc*16 + l%16][kc*32 + 8*(l/16) + {0..7, 16..23}].
static __device__ __forceinline__ v16bf load_b(const __bf16* __restrict__ pack,
                                               int jc, int kc, int lane) {
  return *(const v16bf*)(pack + (size_t)(((jc << 3) + kc) * 32 + lane) * 16);
}

// ---------------------------------------------------------------------------
// Pack W[R][C] (f32 row-major, C==256) -> WMMA-B-native bf16 layout.
// dst[((jc*8 + kc)*32 + l)*16 + n] = bf16(W[jc*16+(l&15)][kc*32 + 8*(l>>4) + (n&7) + 16*(n>>3)])
// ---------------------------------------------------------------------------
__global__ void __launch_bounds__(256) pack_w_kernel(const float* __restrict__ src,
                                                     __bf16* __restrict__ dst,
                                                     int R, int C) {
  int e = blockIdx.x * 256 + threadIdx.x;
  int total = R * C;
  if (e >= total) return;
  int n   = e & 15;
  int l   = (e >> 4) & 31;
  int blk = e >> 9;
  int KC  = C >> 5;
  int kc  = blk % KC;
  int jc  = blk / KC;
  dst[e] = (__bf16)src[(size_t)(jc * 16 + (l & 15)) * C + kc * 32 + (n & 7) + 8 * (l >> 4) + 16 * (n >> 3)];
}

// ---------------------------------------------------------------------------
// Fused SAGE-LSTM layer. 128 threads (4 waves), 64 dst nodes per block.
// LAYER==1: writes LN'd 256-dim rows to out (f32).
// LAYER==2: additionally applies BatchNorm + MLP head, writes scalar per node.
// ---------------------------------------------------------------------------
template <int LAYER>
__global__ void __launch_bounds__(128) sage_lstm_layer(
    const float* __restrict__ src,        // [n_src][256] gather + self source
    const int*   __restrict__ idx,        // [n_dst][8]
    const __bf16* __restrict__ p_ih, const __bf16* __restrict__ p_hh,     // packed [1024][256]
    const __bf16* __restrict__ p_self, const __bf16* __restrict__ p_neigh,// packed [256][256]
    const float* __restrict__ b_ih, const float* __restrict__ b_hh,
    const float* __restrict__ b_neigh,
    const float* __restrict__ ln_g, const float* __restrict__ ln_b,
    const float* __restrict__ bn_g, const float* __restrict__ bn_b,
    const float* __restrict__ bn_mean, const float* __restrict__ bn_var,
    const __bf16* __restrict__ p_w1, const float* __restrict__ b1,
    const float* __restrict__ w2, const float* __restrict__ b2,
    float* __restrict__ out, int n_dst) {
  extern __shared__ char smem[];
  __bf16* xt   = (__bf16*)smem;                                  // 64*256 bf16 (32 KB)
  __bf16* hb   = (__bf16*)(smem + 32768);                        // 2 * 64*256 bf16 (64 KB)
  float*  tmp  = (float*)(smem + 32768 + 65536);                 // 64*256 f32 (c, then pre-LN)
  float*  bsum = (float*)(smem + 32768 + 65536 + 65536);         // 1024 f32 combined bias
  int*    sidx = (int*)(smem + 32768 + 65536 + 65536 + 4096);    // 64 ints

  const int tid  = threadIdx.x;
  const int lane = tid & 31;
  const int wid  = tid >> 5;
  const int tile = wid * 16;          // this wave's 16-node tile
  const int n    = lane & 15;         // C/D column for this lane
  const int hi   = lane >> 4;         // C/D row-half selector
  const int arow = tile + n;          // A-matrix row for this lane
  const int node0 = blockIdx.x * 64;
  if (node0 >= n_dst) return;

  // init: h=0 (buffer 0), c=0, combined LSTM bias
  for (int i = tid; i < 64 * 256; i += 128) { hb[i] = (__bf16)0.0f; tmp[i] = 0.0f; }
  for (int i = tid; i < 1024; i += 128) bsum[i] = b_ih[i] + b_hh[i];
  __syncthreads();

  // ---------------- LSTM over 8 timesteps ----------------
#pragma unroll 1
  for (int t = 0; t < 8; ++t) {
    if (tid < 64) {
      sidx[tid] = idx[(size_t)(node0 + tid) * 8 + t];
      if (t + 1 < 8) {  // prefetch next timestep's neighbor rows
        int nx = idx[(size_t)(node0 + tid) * 8 + t + 1];
        __builtin_prefetch(src + (size_t)nx * 256, 0, 0);
      }
    }
    __syncthreads();
    // gather x_t (64 rows x 256 f32) -> bf16 LDS
    for (int i = tid; i < 64 * 64; i += 128) {
      int row = i >> 6, c4 = i & 63;
      const float4 f = ((const float4*)(src + (size_t)sidx[row] * 256))[c4];
      __bf16* d = xt + row * 256 + c4 * 4;
      d[0] = (__bf16)f.x; d[1] = (__bf16)f.y; d[2] = (__bf16)f.z; d[3] = (__bf16)f.w;
    }
    __syncthreads();

    const __bf16* hc = hb + (size_t)(t & 1) * 64 * 256;
    __bf16*       hn = hb + (size_t)((t + 1) & 1) * 64 * 256;

    // hoist A fragments for x_t and h (reused across all 16 output chunks)
    v16bf ax[8], ah[8];
#pragma unroll
    for (int kc = 0; kc < 8; ++kc) {
      ax[kc] = load_a(xt, arow, kc, hi);
      ah[kc] = load_a(hc, arow, kc, hi);
    }

#pragma unroll 1
    for (int jj = 0; jj < 16; ++jj) {
      v8f zi = {}, zf = {}, zg = {}, zo = {};
#pragma unroll
      for (int kc = 0; kc < 8; ++kc) {
        zi = WMMA_BF16(ax[kc], load_b(p_ih, jj,      kc, lane), zi);
        zi = WMMA_BF16(ah[kc], load_b(p_hh, jj,      kc, lane), zi);
        zf = WMMA_BF16(ax[kc], load_b(p_ih, 16 + jj, kc, lane), zf);
        zf = WMMA_BF16(ah[kc], load_b(p_hh, 16 + jj, kc, lane), zf);
        zg = WMMA_BF16(ax[kc], load_b(p_ih, 32 + jj, kc, lane), zg);
        zg = WMMA_BF16(ah[kc], load_b(p_hh, 32 + jj, kc, lane), zg);
        zo = WMMA_BF16(ax[kc], load_b(p_ih, 48 + jj, kc, lane), zo);
        zo = WMMA_BF16(ah[kc], load_b(p_hh, 48 + jj, kc, lane), zo);
      }
      const int col = jj * 16 + n;
      const float bi = bsum[col], bf_ = bsum[256 + col], bg = bsum[512 + col], bo = bsum[768 + col];
#pragma unroll
      for (int v = 0; v < 8; ++v) {
        int off = (tile + v + 8 * hi) * 256 + col;
        float iv = sigf(zi[v] + bi);
        float fv = sigf(zf[v] + bf_);
        float gv = tanhf(zg[v] + bg);
        float ov = sigf(zo[v] + bo);
        float cn = fv * tmp[off] + iv * gv;   // c in LDS (f32)
        tmp[off] = cn;
        hn[off]  = (__bf16)(ov * tanhf(cn));
      }
    }
    __syncthreads();
  }
  // final h is in buffer 0 (t==8 even)

  // ---------------- SAGE: gelu(h_self W_self^T + h_n W_neigh^T + b) ---------
  for (int i = tid; i < 64 * 64; i += 128) {  // stage h_self -> xt (bf16)
    int row = i >> 6, c4 = i & 63;
    const float4 f = ((const float4*)(src + (size_t)(node0 + row) * 256))[c4];
    __bf16* d = xt + row * 256 + c4 * 4;
    d[0] = (__bf16)f.x; d[1] = (__bf16)f.y; d[2] = (__bf16)f.z; d[3] = (__bf16)f.w;
  }
  __syncthreads();

  {
    const __bf16* hf = hb;  // final h
    v16bf as[8], an[8];
#pragma unroll
    for (int kc = 0; kc < 8; ++kc) {
      as[kc] = load_a(xt, arow, kc, hi);
      an[kc] = load_a(hf, arow, kc, hi);
    }
#pragma unroll 1
    for (int jj = 0; jj < 16; ++jj) {
      v8f acc = {};
#pragma unroll
      for (int kc = 0; kc < 8; ++kc) {
        acc = WMMA_BF16(as[kc], load_b(p_self,  jj, kc, lane), acc);
        acc = WMMA_BF16(an[kc], load_b(p_neigh, jj, kc, lane), acc);
      }
      const int col = jj * 16 + n;
      const float bb = b_neigh[col];
#pragma unroll
      for (int v = 0; v < 8; ++v)
        tmp[(tile + v + 8 * hi) * 256 + col] = geluf(acc[v] + bb);
    }
  }
  __syncthreads();

  // ---------------- LayerNorm (wave-local rows) -----------------------------
  for (int r = tile; r < tile + 16; ++r) {
    float s = 0.0f, s2 = 0.0f;
    for (int c = lane; c < 256; c += 32) {
      float x = tmp[r * 256 + c];
      s += x; s2 += x * x;
    }
#pragma unroll
    for (int off = 16; off >= 1; off >>= 1) {
      s  += __shfl_xor(s,  off, 32);
      s2 += __shfl_xor(s2, off, 32);
    }
    float mu   = s * (1.0f / 256.0f);
    float var  = s2 * (1.0f / 256.0f) - mu * mu;
    float rstd = rsqrtf(var + 1e-5f);
    for (int c = lane; c < 256; c += 32) {
      float y = (tmp[r * 256 + c] - mu) * rstd * ln_g[c] + ln_b[c];
      if (LAYER == 1) {
        out[(size_t)(node0 + r) * 256 + c] = y;
      } else {
        float yb = (y - bn_mean[c]) * rsqrtf(bn_var[c] + 1e-5f) * bn_g[c] + bn_b[c];
        xt[r * 256 + c] = (__bf16)yb;  // MLP input
      }
    }
  }

  // ---------------- Layer-2 head: gelu(h W1^T + b1) @ W2^T + b2 -------------
  if (LAYER == 2) {
    __syncthreads();
    v16bf am[8];
#pragma unroll
    for (int kc = 0; kc < 8; ++kc) am[kc] = load_a(xt, arow, kc, hi);
    __bf16* mout = hb;  // reuse (h no longer needed)
#pragma unroll 1
    for (int jj = 0; jj < 16; ++jj) {
      v8f acc = {};
#pragma unroll
      for (int kc = 0; kc < 8; ++kc)
        acc = WMMA_BF16(am[kc], load_b(p_w1, jj, kc, lane), acc);
      const int col = jj * 16 + n;
      const float bb = b1[col];
#pragma unroll
      for (int v = 0; v < 8; ++v)
        mout[(tile + v + 8 * hi) * 256 + col] = (__bf16)geluf(acc[v] + bb);
    }
    __syncthreads();
    if (tid < 64) {
      float sAcc = b2[0];
      for (int c = 0; c < 256; ++c) sAcc += (float)mout[tid * 256 + c] * w2[c];
      out[node0 + tid] = sAcc;
    }
  }
}

// ---------------------------------------------------------------------------
// Host launcher
// ---------------------------------------------------------------------------
extern "C" void kernel_launch(void* const* d_in, const int* in_sizes, int n_in,
                              void* d_out, int out_size, void* d_ws, size_t ws_size,
                              hipStream_t stream) {
  (void)in_sizes; (void)n_in; (void)out_size; (void)ws_size;
  const float* h0       = (const float*)d_in[0];
  const int*   idx0     = (const int*)  d_in[1];
  const int*   idx1     = (const int*)  d_in[2];
  const float* W_ih1    = (const float*)d_in[3];
  const float* W_hh1    = (const float*)d_in[4];
  const float* b_ih1    = (const float*)d_in[5];
  const float* b_hh1    = (const float*)d_in[6];
  const float* W_self1  = (const float*)d_in[7];
  const float* W_neigh1 = (const float*)d_in[8];
  const float* b_neigh1 = (const float*)d_in[9];
  const float* W_ih2    = (const float*)d_in[10];
  const float* W_hh2    = (const float*)d_in[11];
  const float* b_ih2    = (const float*)d_in[12];
  const float* b_hh2    = (const float*)d_in[13];
  const float* W_self2  = (const float*)d_in[14];
  const float* W_neigh2 = (const float*)d_in[15];
  const float* b_neigh2 = (const float*)d_in[16];
  const float* ln_g     = (const float*)d_in[17];
  const float* ln_b     = (const float*)d_in[18];
  const float* bn_g     = (const float*)d_in[19];
  const float* bn_b     = (const float*)d_in[20];
  const float* bn_mean  = (const float*)d_in[21];
  const float* bn_var   = (const float*)d_in[22];
  const float* W1       = (const float*)d_in[23];
  const float* b1       = (const float*)d_in[24];
  const float* W2       = (const float*)d_in[25];
  const float* b2       = (const float*)d_in[26];

  char* ws = (char*)d_ws;
  size_t off = 0;
  float* l1out = (float*)(ws + off); off += (size_t)16384 * 256 * 4;
  auto take = [&](size_t bytes) { void* p = ws + off; off += bytes; return p; };
  __bf16* p_ih1    = (__bf16*)take((size_t)1024 * 256 * 2);
  __bf16* p_hh1    = (__bf16*)take((size_t)1024 * 256 * 2);
  __bf16* p_self1  = (__bf16*)take((size_t)256 * 256 * 2);
  __bf16* p_neigh1 = (__bf16*)take((size_t)256 * 256 * 2);
  __bf16* p_ih2    = (__bf16*)take((size_t)1024 * 256 * 2);
  __bf16* p_hh2    = (__bf16*)take((size_t)1024 * 256 * 2);
  __bf16* p_self2  = (__bf16*)take((size_t)256 * 256 * 2);
  __bf16* p_neigh2 = (__bf16*)take((size_t)256 * 256 * 2);
  __bf16* p_w1     = (__bf16*)take((size_t)256 * 256 * 2);

  const int big = (1024 * 256) / 256, small = (256 * 256) / 256;
  pack_w_kernel<<<big,   256, 0, stream>>>(W_ih1,    p_ih1,    1024, 256);
  pack_w_kernel<<<big,   256, 0, stream>>>(W_hh1,    p_hh1,    1024, 256);
  pack_w_kernel<<<small, 256, 0, stream>>>(W_self1,  p_self1,  256,  256);
  pack_w_kernel<<<small, 256, 0, stream>>>(W_neigh1, p_neigh1, 256,  256);
  pack_w_kernel<<<big,   256, 0, stream>>>(W_ih2,    p_ih2,    1024, 256);
  pack_w_kernel<<<big,   256, 0, stream>>>(W_hh2,    p_hh2,    1024, 256);
  pack_w_kernel<<<small, 256, 0, stream>>>(W_self2,  p_self2,  256,  256);
  pack_w_kernel<<<small, 256, 0, stream>>>(W_neigh2, p_neigh2, 256,  256);
  pack_w_kernel<<<small, 256, 0, stream>>>(W1,       p_w1,     256,  256);

  const size_t smem = 32768 + 65536 + 65536 + 4096 + 256;
  sage_lstm_layer<1><<<16384 / 64, 128, smem, stream>>>(
      h0, idx0, p_ih1, p_hh1, p_self1, p_neigh1, b_ih1, b_hh1, b_neigh1,
      ln_g, ln_b, nullptr, nullptr, nullptr, nullptr, nullptr, nullptr,
      nullptr, nullptr, l1out, 16384);
  sage_lstm_layer<2><<<4096 / 64, 128, smem, stream>>>(
      l1out, idx1, p_ih2, p_hh2, p_self2, p_neigh2, b_ih2, b_hh2, b_neigh2,
      ln_g, ln_b, bn_g, bn_b, bn_mean, bn_var, p_w1, b1, W2, b2,
      (float*)d_out, 4096);
}
